// Attention_8735963480317
// MI455X (gfx1250) — compile-verified, matrix operand order checked by reading, not run on hardware
//
#include <hip/hip_runtime.h>
#include <hip/hip_bf16.h>

// Problem sizes (fixed by the reference)
#define SRC   1024          // S
#define NB    64            // N (batch)
#define HH    1024          // H == INPUT_SIZE
#define RROWS (SRC * NB)    // 65536 flattened rows of out_e

// ---- WMMA types (CDNA5 gfx1250, wave32) ----
typedef __attribute__((ext_vector_type(16))) __bf16 v16bf;
typedef __attribute__((ext_vector_type(8)))  float  v8f;

union Frag16 {           // 32 bytes: one bf16 A/B operand (16 bf16 per lane)
    v16bf v;
    uint4 q[2];
};

// float -> bf16 bits, round-to-nearest-even
__device__ __forceinline__ unsigned short f2bf(float f) {
    union { float f; unsigned u; } x; x.f = f;
    unsigned r = x.u + 0x7FFFu + ((x.u >> 16) & 1u);
    return (unsigned short)(r >> 16);
}

// ---------------------------------------------------------------------------
// Kernel 0: convert W2 [1024,1024] f32 -> bf16 (row j == column j of W2.T)
// ---------------------------------------------------------------------------
__global__ __launch_bounds__(256) void k_convert_w2(const float* __restrict__ W2,
                                                    unsigned short* __restrict__ W2bf) {
    int base = (blockIdx.x * 256 + threadIdx.x) * 4;
    float4 f = *(const float4*)(W2 + base);
    W2bf[base + 0] = f2bf(f.x);
    W2bf[base + 1] = f2bf(f.y);
    W2bf[base + 2] = f2bf(f.z);
    W2bf[base + 3] = f2bf(f.w);
}

// ---------------------------------------------------------------------------
// Kernel 1: t1[n][j] = b1[j] + b2[j] + sum_h hidden_d[n][h] * W1[j][h]
// One block per output feature j; W1 row staged in LDS; 4 threads per n.
// ---------------------------------------------------------------------------
__global__ __launch_bounds__(256) void k_hidden_proj(const float* __restrict__ hd,
                                                     const float* __restrict__ W1,
                                                     const float* __restrict__ b1,
                                                     const float* __restrict__ b2,
                                                     float* __restrict__ t1) {
    const int j = blockIdx.x;
    __shared__ float wrow[HH];
    for (int i = threadIdx.x; i < HH; i += 256)
        wrow[i] = W1[(size_t)j * HH + i];
    __syncthreads();

    const int n = threadIdx.x >> 2;   // 0..63
    const int q = threadIdx.x & 3;    // quarter of H
    const float* h = hd + (size_t)n * HH + q * 256;
    const float* w = wrow + q * 256;
    float s = 0.f;
#pragma unroll 4
    for (int i = 0; i < 256; ++i) s += h[i] * w[i];
    s += __shfl_xor(s, 1, 32);
    s += __shfl_xor(s, 2, 32);
    if (q == 0) t1[(size_t)n * HH + j] = s + b1[j] + b2[j];
}

// ---------------------------------------------------------------------------
// Kernel 2 (main): fused  a[s,n] = W3 . tanh(out_e@W2.T + t1[n]) + b3
// Block = 256 threads (8 waves). Block tile: 32 rows x 1024 cols.
// Wave w: 2 row-tiles x 8 col-tiles (cols w*128 .. w*128+127).
// bf16 WMMA (v_wmma_f32_16x16x32_bf16), f32 accumulate, epilogue fused so the
// 256MB intermediate "e" never touches memory.
// ---------------------------------------------------------------------------
__global__ __launch_bounds__(256) void k_fused_gemm(const float* __restrict__ out_e,
                                                    const unsigned short* __restrict__ W2bf,
                                                    const float* __restrict__ t1,
                                                    const float* __restrict__ W3,
                                                    const float* __restrict__ b3,
                                                    float* __restrict__ a_ws) {
    __shared__ __align__(16) unsigned short As[32 * 40]; // 32 rows, K=32 bf16, +8 pad
    __shared__ float a_red[32];

    const int tid  = threadIdx.x;
    const int lane = tid & 31;
    const int wave = tid >> 5;
    const int r0   = blockIdx.x * 32;        // first of 32 rows
    const int colw = wave * 128;             // this wave's first column

    const int mcol = lane & 15;              // N / M-in-tile index
    const int kb   = (lane >> 4) * 8;        // half-wave K base (0 or 8)
    const int roff = (lane >> 4) * 8;        // C rows: element r -> M = roff + r

    v8f acc[2][8] = {};                      // 2 row-tiles x 8 col-tiles

    for (int k0 = 0; k0 < HH; k0 += 32) {
        // --- cooperative A stage: 32 rows x 32 k, f32 -> bf16 into LDS ---
        {
            int e   = tid * 4;               // 1024 elements / 256 threads
            int row = e >> 5;
            int kk  = e & 31;
            float4 f = *(const float4*)(out_e + (size_t)(r0 + row) * HH + k0 + kk);
            unsigned short* dst = &As[row * 40 + kk];
            dst[0] = f2bf(f.x); dst[1] = f2bf(f.y);
            dst[2] = f2bf(f.z); dst[3] = f2bf(f.w);
        }
        __syncthreads();

        // --- A fragments (one per row-tile) from LDS: ds_load_b128 x2 ---
        Frag16 a[2];
#pragma unroll
        for (int u = 0; u < 2; ++u) {
            const unsigned short* base = &As[(u * 16 + mcol) * 40 + kb];
            a[u].q[0] = *(const uint4*)(base);        // K = kb .. kb+7
            a[u].q[1] = *(const uint4*)(base + 16);   // K = kb+16 .. kb+23
        }

        // --- B fragments + WMMA per column tile ---
#pragma unroll
        for (int t = 0; t < 8; ++t) {
            const int col = colw + t * 16 + mcol;     // W2.T column == W2 row
            const unsigned short* wr = W2bf + (size_t)col * HH + k0 + kb;
            Frag16 b;
            b.q[0] = *(const uint4*)(wr);
            b.q[1] = *(const uint4*)(wr + 16);
            acc[0][t] = __builtin_amdgcn_wmma_f32_16x16x32_bf16(
                false, a[0].v, false, b.v, (short)0, acc[0][t], false, false);
            acc[1][t] = __builtin_amdgcn_wmma_f32_16x16x32_bf16(
                false, a[1].v, false, b.v, (short)0, acc[1][t], false, false);
        }
        __syncthreads();
    }

    // --- fused epilogue: tanh, dot with W3, reduce to a[s,n] ---
    if (tid < 32) a_red[tid] = 0.f;
    __syncthreads();

#pragma unroll
    for (int u = 0; u < 2; ++u) {
        float part[8];
#pragma unroll
        for (int r = 0; r < 8; ++r) part[r] = 0.f;
#pragma unroll
        for (int t = 0; t < 8; ++t) {
            const int col = colw + t * 16 + mcol;
            const float w3 = W3[col];
#pragma unroll
            for (int r = 0; r < 8; ++r) {
                const int grow = r0 + u * 16 + roff + r;     // flat row = s*N+n
                const int n = grow & (NB - 1);
                float e = tanhf(acc[u][t][r] + t1[(size_t)n * HH + col]);
                part[r] += e * w3;
            }
        }
        // reduce across the 16 lanes of each half-wave (rows identical there)
#pragma unroll
        for (int m = 1; m < 16; m <<= 1)
#pragma unroll
            for (int r = 0; r < 8; ++r)
                part[r] += __shfl_xor(part[r], m, 32);
        if ((lane & 15) == 0) {
#pragma unroll
            for (int r = 0; r < 8; ++r)
                atomicAdd(&a_red[u * 16 + roff + r], part[r]);
        }
    }
    __syncthreads();
    if (tid < 32) a_ws[r0 + tid] = a_red[tid] + b3[0];
}

// ---------------------------------------------------------------------------
// Kernel 3: softmax over the source dim (S=1024) per batch n; writes alpha
// directly into its slot of d_out (layout [S,N,1] flat = s*N+n).
// ---------------------------------------------------------------------------
__global__ __launch_bounds__(256) void k_softmax(const float* __restrict__ a_ws,
                                                 float* __restrict__ alpha) {
    const int n = blockIdx.x, tid = threadIdx.x;
    __shared__ float red[256];
    float v[4];
    float mx = -3.4e38f;
#pragma unroll
    for (int i = 0; i < 4; ++i) {
        v[i] = a_ws[(size_t)(tid + i * 256) * NB + n];
        mx = fmaxf(mx, v[i]);
    }
    red[tid] = mx; __syncthreads();
    for (int st = 128; st > 0; st >>= 1) {
        if (tid < st) red[tid] = fmaxf(red[tid], red[tid + st]);
        __syncthreads();
    }
    mx = red[0]; __syncthreads();

    float sum = 0.f;
#pragma unroll
    for (int i = 0; i < 4; ++i) { v[i] = __expf(v[i] - mx); sum += v[i]; }
    red[tid] = sum; __syncthreads();
    for (int st = 128; st > 0; st >>= 1) {
        if (tid < st) red[tid] += red[tid + st];
        __syncthreads();
    }
    const float inv = 1.0f / red[0];
#pragma unroll
    for (int i = 0; i < 4; ++i)
        alpha[(size_t)(tid + i * 256) * NB + n] = v[i] * inv;
}

// ---------------------------------------------------------------------------
// Kernel 4: c[n,h] = sum_s alpha[s,n] * out_e[s,n,h]  (streaming, HBM-bound)
// ---------------------------------------------------------------------------
__global__ __launch_bounds__(256) void k_context(const float* __restrict__ out_e,
                                                 const float* __restrict__ alpha,
                                                 float* __restrict__ c) {
    const int idx = blockIdx.x * 256 + threadIdx.x;   // 0 .. 65535 == n*1024+h
    const int n = idx >> 10;
    const float* p = out_e + idx;                     // s=0 element of (n,h)
    float s0 = 0.f, s1 = 0.f, s2 = 0.f, s3 = 0.f;
    for (int s = 0; s < SRC; s += 4) {
        s0 += alpha[(size_t)(s + 0) * NB + n] * p[(size_t)(s + 0) * (NB * HH)];
        s1 += alpha[(size_t)(s + 1) * NB + n] * p[(size_t)(s + 1) * (NB * HH)];
        s2 += alpha[(size_t)(s + 2) * NB + n] * p[(size_t)(s + 2) * (NB * HH)];
        s3 += alpha[(size_t)(s + 3) * NB + n] * p[(size_t)(s + 3) * (NB * HH)];
    }
    c[idx] = (s0 + s1) + (s2 + s3);
}

// ---------------------------------------------------------------------------
extern "C" void kernel_launch(void* const* d_in, const int* in_sizes, int n_in,
                              void* d_out, int out_size, void* d_ws, size_t ws_size,
                              hipStream_t stream) {
    (void)in_sizes; (void)n_in; (void)out_size; (void)ws_size;
    const float* out_e    = (const float*)d_in[0];
    const float* hidden_d = (const float*)d_in[1];
    const float* W1       = (const float*)d_in[2];
    const float* b1       = (const float*)d_in[3];
    const float* W2       = (const float*)d_in[4];
    const float* b2       = (const float*)d_in[5];
    const float* W3       = (const float*)d_in[6];
    const float* b3       = (const float*)d_in[7];

    float* c_out     = (float*)d_out;                 // [1,N,H] = 65536 f32
    float* alpha_out = c_out + (size_t)NB * HH;       // [S,N,1] = 65536 f32

    // workspace layout
    unsigned short* W2bf = (unsigned short*)d_ws;                         // 2 MB
    float* t1   = (float*)((char*)d_ws + 2u * 1024u * 1024u);             // 256 KB
    float* a_ws = (float*)((char*)d_ws + 2u * 1024u * 1024u + 256u * 1024u); // 256 KB

    k_convert_w2 <<<1024, 256, 0, stream>>>(W2, W2bf);
    k_hidden_proj<<<1024, 256, 0, stream>>>(hidden_d, W1, b1, b2, t1);
    k_fused_gemm <<<RROWS / 32, 256, 0, stream>>>(out_e, W2bf, t1, W3, b3, a_ws);
    k_softmax    <<<NB, 256, 0, stream>>>(a_ws, alpha_out);
    k_context    <<<(NB * HH) / 256, 256, 0, stream>>>(out_e, alpha_out, c_out);
}
